// QKV_18743237280620
// MI455X (gfx1250) — matrix operand order repaired and checked by reading, not compiled, not run on hardware
//
#include <hip/hip_runtime.h>
#include <stdint.h>

// ---- problem constants (match reference) ----
constexpr int BB = 4;
constexpr int SS = 2048;
constexpr int EE = 1024;
constexpr int HH = 16;
constexpr int DD = 64;
constexpr int MM = BB * SS;          // 8192 rows
constexpr float EPSV = 1e-5f;
constexpr float NEGV = -1e9f;

typedef __attribute__((ext_vector_type(16))) __bf16 v16bf;
typedef __attribute__((ext_vector_type(8)))  float  v8f;

union Frag { v16bf v; uint4 q[2]; unsigned short u[16]; };

__device__ inline unsigned short f2bf(float x) {
  unsigned int u = __float_as_uint(x);
  unsigned int r = u + 0x7FFFu + ((u >> 16) & 1u);   // round-to-nearest-even
  return (unsigned short)(r >> 16);
}

// gfx1250 async global->LDS copies (ASYNCcnt-tracked, no VGPR round trip).
__device__ inline void async_b128(unsigned lds_addr, uint64_t gaddr) {
  asm volatile("global_load_async_to_lds_b128 %0, %1, off"
               :: "v"(lds_addr), "v"(gaddr) : "memory");
}
__device__ inline void async_b128x4(unsigned lds_addr, uint64_t gaddr) {
  asm volatile("global_load_async_to_lds_b128 %0, %1, off\n\t"
               "global_load_async_to_lds_b128 %0, %1, off offset:16\n\t"
               "global_load_async_to_lds_b128 %0, %1, off offset:32\n\t"
               "global_load_async_to_lds_b128 %0, %1, off offset:48"
               :: "v"(lds_addr), "v"(gaddr) : "memory");
}
__device__ inline void wait_async0() {
  asm volatile("s_wait_asynccnt 0x0" ::: "memory");
}

// ---------------- fp32 -> bf16 elementwise ----------------
__global__ void cvt_bf16_kernel(const float* __restrict__ in,
                                unsigned short* __restrict__ out, int n) {
  int i = blockIdx.x * blockDim.x + threadIdx.x;
  if (i < n) out[i] = f2bf(in[i]);
}

// ---------------- fp32 W (KxN) -> bf16 W^T (NxK), 32x32 LDS tile ----------------
__global__ __launch_bounds__(256)
void cvt_wT_kernel(const float* __restrict__ in, unsigned short* __restrict__ out) {
  __shared__ float tile[32][33];
  const int tx = threadIdx.x & 31;       // 32 x 8 logical threads
  const int ty = threadIdx.x >> 5;
  const int n0 = blockIdx.x * 32;
  const int k0 = blockIdx.y * 32;
  for (int i = 0; i < 4; i++) {
    int kr = ty + i * 8;
    tile[kr][tx] = in[(size_t)(k0 + kr) * EE + n0 + tx];
  }
  __syncthreads();
  for (int i = 0; i < 4; i++) {
    int nr = ty + i * 8;
    out[(size_t)(n0 + nr) * EE + k0 + tx] = f2bf(tile[tx][nr]);
  }
}

// ---------------- LayerNorm over E=1024, one row per block ----------------
__global__ __launch_bounds__(256)
void ln_kernel(const float* __restrict__ x, const float* __restrict__ gamma,
               const float* __restrict__ beta, float* __restrict__ outf,
               unsigned short* __restrict__ outbf) {
  const int row = blockIdx.x;
  const int tid = threadIdx.x;
  const float* xr = x + (size_t)row * EE;
  float v[4];
  float s = 0.f, s2 = 0.f;
  for (int i = 0; i < 4; i++) {
    float t = xr[tid + 256 * i];
    v[i] = t; s += t; s2 += t * t;
  }
  __shared__ float rs[256], rq[256];
  rs[tid] = s; rq[tid] = s2;
  __syncthreads();
  for (int st = 128; st > 0; st >>= 1) {
    if (tid < st) { rs[tid] += rs[tid + st]; rq[tid] += rq[tid + st]; }
    __syncthreads();
  }
  const float mean = rs[0] * (1.0f / EE);
  const float var  = rq[0] * (1.0f / EE) - mean * mean;
  const float inv  = rsqrtf(var + EPSV);
  for (int i = 0; i < 4; i++) {
    int col = tid + 256 * i;
    float y = (v[i] - mean) * inv * gamma[col] + beta[col];
    size_t idx = (size_t)row * EE + col;
    outf[idx] = y;
    if (outbf) outbf[idx] = f2bf(y);
  }
}

// ---------------- tiled bf16 WMMA GEMM: C = A(MxK) * Wt^T + bias ----------------
// Block tile 128x128 (4 waves, 2x2, 64x64 per wave).  16 WMMA / 16 ds_load per
// wave per K-step.  Double-buffered async-to-LDS staging.
// A: MxK bf16 row-major.  Wt: NxK bf16 row-major (pre-transposed weights).
// mode 0: bf16 out scattered to [B,H,S,D]   (q/k heads)
// mode 2: bf16 out scattered to [B,H,D,S]   (v heads, pre-transposed for flash)
// mode 1: f32 out = acc + bias + residual   (final projection)
__global__ __launch_bounds__(128)
void gemm_kernel(const unsigned short* __restrict__ A,
                 const unsigned short* __restrict__ Wt,
                 const float* __restrict__ bias,
                 int K, int N, int mode,
                 unsigned short* __restrict__ outb,
                 float* __restrict__ outf,
                 const float* __restrict__ resid) {
  constexpr int TM = 128, TN = 128, TK = 32, PAD = 8;
  __shared__ unsigned short As[2][TM][TK + PAD];   // [m][k]
  __shared__ unsigned short Bs[2][TN][TK + PAD];   // [n][k]
  const int tid  = threadIdx.x;
  const int lane = tid & 31;
  const int wave = tid >> 5;
  const int g    = lane >> 4;       // lane group (0/1)
  const int lr   = lane & 15;
  const int wm0  = (wave & 1) * 64; // wave tile origin within block tile
  const int wn0  = (wave >> 1) * 64;
  const int n0 = blockIdx.x * TN;
  const int m0 = blockIdx.y * TM;

  // async-stage one K-slab: each thread stages one 64B row of A and of Wt
  auto stage = [&](int buf, int kk) {
    async_b128x4((unsigned)(uintptr_t)&As[buf][tid][0],
                 (uint64_t)(uintptr_t)(A  + (size_t)(m0 + tid) * K + kk));
    async_b128x4((unsigned)(uintptr_t)&Bs[buf][tid][0],
                 (uint64_t)(uintptr_t)(Wt + (size_t)(n0 + tid) * K + kk));
  };

  v8f acc[4][4];
  for (int mt = 0; mt < 4; mt++)
    for (int nt = 0; nt < 4; nt++)
      for (int r = 0; r < 8; r++) acc[mt][nt][r] = 0.f;

  stage(0, 0);
  wait_async0();
  __syncthreads();

  for (int kk = 0; kk < K; kk += TK) {
    const int cur = (kk / TK) & 1;
    if (kk + TK < K) stage(cur ^ 1, kk + TK);   // overlap fill with compute

    Frag af[4];
    for (int mt = 0; mt < 4; mt++) {
      const int am = wm0 + mt * 16 + lr;
      af[mt].q[0] = *(const uint4*)&As[cur][am][8 * g];
      af[mt].q[1] = *(const uint4*)&As[cur][am][16 + 8 * g];
    }
    for (int nt = 0; nt < 4; nt++) {
      Frag bf;
      const int bn = wn0 + nt * 16 + lr;
      bf.q[0] = *(const uint4*)&Bs[cur][bn][8 * g];
      bf.q[1] = *(const uint4*)&Bs[cur][bn][16 + 8 * g];
      for (int mt = 0; mt < 4; mt++)
        acc[mt][nt] = __builtin_amdgcn_wmma_f32_16x16x32_bf16(
            false, af[mt].v, false, bf.v, (short)0, acc[mt][nt], false, false);
    }
    wait_async0();
    __syncthreads();
  }

  // store (C layout: VGPR r -> M = r + 8*g, lane%16 -> N)
  for (int mt = 0; mt < 4; mt++) {
    for (int nt = 0; nt < 4; nt++) {
      const int nG = n0 + wn0 + nt * 16 + lr;
      const float bs = bias[nG];
      for (int r = 0; r < 8; r++) {
        const int mG = m0 + wm0 + mt * 16 + r + 8 * g;
        float val = acc[mt][nt][r] + bs;
        if (mode == 1) {
          size_t idx = (size_t)mG * N + nG;
          outf[idx] = val + resid[idx];
        } else {
          int bb = mG / SS, ss = mG % SS;
          int hh = nG / DD, dd = nG % DD;
          size_t idx = (mode == 0)
              ? ((size_t)(bb * HH + hh) * SS + ss) * DD + dd    // [B,H,S,D]
              : ((size_t)(bb * HH + hh) * DD + dd) * SS + ss;   // [B,H,D,S]
          outb[idx] = f2bf(val);
        }
      }
    }
  }
}

// ---------------- flash attention: per (b,h,64-query tile) ----------------
// qh/kh: [B,H,S,D] bf16;  vhT: [B,H,D,S] bf16 (pre-transposed V heads)
__global__ __launch_bounds__(128)
void flash_kernel(const unsigned short* __restrict__ qh,
                  const unsigned short* __restrict__ kh,
                  const unsigned short* __restrict__ vhT,
                  const unsigned char* __restrict__ mask,
                  unsigned short* __restrict__ attn_out) {
  __shared__ unsigned short Kt[2][64][64];     // [key][d]
  __shared__ unsigned short Vt[2][64][64];     // [d][key]
  __shared__ unsigned short Pt[4][16][72];     // per-wave P tile
  const int tid  = threadIdx.x;
  const int lane = tid & 31;
  const int wave = tid >> 5;
  const int g    = lane >> 4;
  const int lr   = lane & 15;
  const int q0 = blockIdx.x * 64;
  const int h  = blockIdx.y;
  const int b  = blockIdx.z;
  const size_t headBase = (size_t)(b * HH + h) * SS * DD;   // same for [S,D] and [D,S]

  // async-stage K tile (contiguous 8KB) and V tile (64 rows x 128B) for block kb
  auto stage_kv = [&](int buf, int kb) {
    const unsigned short* ksrc = kh + headBase + (size_t)kb * 64 * DD;
    const unsigned kbase = (unsigned)(uintptr_t)&Kt[buf][0][0];
    for (int i = tid; i < 512; i += 128)
      async_b128(kbase + i * 16, (uint64_t)(uintptr_t)(ksrc + i * 8));
    const unsigned short* vsrc = vhT + headBase + (size_t)kb * 64;   // row d stride S
    const unsigned vbase = (unsigned)(uintptr_t)&Vt[buf][0][0];
    for (int i = tid; i < 512; i += 128) {
      int d = i >> 3, c = i & 7;                  // 8 x 16B chunks per row
      async_b128(vbase + (unsigned)(d * 64 + c * 8) * 2,
                 (uint64_t)(uintptr_t)(vsrc + (size_t)d * SS + c * 8));
    }
  };

  // Q fragments held in registers for the whole KV loop (K-dim = D = 64 -> 2 steps)
  Frag qa[2];
  {
    const unsigned short* qrow = qh + headBase + (size_t)(q0 + wave * 16 + lr) * DD;
    for (int ks = 0; ks < 2; ks++) {
      qa[ks].q[0] = *(const uint4*)(qrow + 32 * ks + 8 * g);
      qa[ks].q[1] = *(const uint4*)(qrow + 32 * ks + 16 + 8 * g);
    }
  }

  float mrun[8], lrun[8];
  v8f o[4];
  for (int r = 0; r < 8; r++) { mrun[r] = -3.0e38f; lrun[r] = 0.f; }
  for (int nt = 0; nt < 4; nt++)
    for (int r = 0; r < 8; r++) o[nt][r] = 0.f;

  stage_kv(0, 0);
  wait_async0();
  __syncthreads();

  for (int kb = 0; kb < SS / 64; kb++) {
    const int cur = kb & 1;
    if (kb + 1 < SS / 64) stage_kv(cur ^ 1, kb + 1);   // overlap fill with compute

    // scores S = Q * K^T  (B fragment: lane%16 = key, K-dim = d)
    v8f s[4];
    for (int nt = 0; nt < 4; nt++)
      for (int r = 0; r < 8; r++) s[nt][r] = 0.f;
    for (int ks = 0; ks < 2; ks++) {
      for (int nt = 0; nt < 4; nt++) {
        Frag bf;
        const int key = nt * 16 + lr;
        bf.q[0] = *(const uint4*)&Kt[cur][key][32 * ks + 8 * g];
        bf.q[1] = *(const uint4*)&Kt[cur][key][32 * ks + 16 + 8 * g];
        s[nt] = __builtin_amdgcn_wmma_f32_16x16x32_bf16(
            false, qa[ks].v, false, bf.v, (short)0, s[nt], false, false);
      }
    }

    // mask + online softmax (row reductions across 16-lane groups)
    const unsigned char* mrow =
        mask + ((size_t)b * SS + q0 + wave * 16) * SS + (size_t)kb * 64;
    for (int r = 0; r < 8; r++) {
      const int m = r + 8 * g;
      float mx = -3.0e38f;
      for (int nt = 0; nt < 4; nt++) {
        const int key = nt * 16 + lr;
        float val = s[nt][r];
        if (mrow[(size_t)m * SS + key]) val = NEGV;
        s[nt][r] = val;
        mx = fmaxf(mx, val);
      }
      for (int d = 1; d < 16; d <<= 1) mx = fmaxf(mx, __shfl_xor(mx, d, 32));
      const float nm = fmaxf(mrun[r], mx);
      const float alpha = __expf(mrun[r] - nm);
      float sum = 0.f;
      for (int nt = 0; nt < 4; nt++) {
        float p = __expf(s[nt][r] - nm);
        s[nt][r] = p; sum += p;
      }
      for (int d = 1; d < 16; d <<= 1) sum += __shfl_xor(sum, d, 32);
      lrun[r] = lrun[r] * alpha + sum;
      mrun[r] = nm;
      for (int nt = 0; nt < 4; nt++) o[nt][r] *= alpha;
      for (int nt = 0; nt < 4; nt++)
        Pt[wave][m][nt * 16 + lr] = f2bf(s[nt][r]);
    }

    // O += P * V  (A = wave-private P tile; B: lane%16 = d, K-dim = key)
    for (int ks = 0; ks < 2; ks++) {
      Frag pa;
      pa.q[0] = *(const uint4*)&Pt[wave][lr][32 * ks + 8 * g];
      pa.q[1] = *(const uint4*)&Pt[wave][lr][32 * ks + 16 + 8 * g];
      for (int nt = 0; nt < 4; nt++) {
        Frag bf;
        const int dd = nt * 16 + lr;
        bf.q[0] = *(const uint4*)&Vt[cur][dd][32 * ks + 8 * g];
        bf.q[1] = *(const uint4*)&Vt[cur][dd][32 * ks + 16 + 8 * g];
        o[nt] = __builtin_amdgcn_wmma_f32_16x16x32_bf16(
            false, pa.v, false, bf.v, (short)0, o[nt], false, false);
      }
    }
    wait_async0();
    __syncthreads();
  }

  // normalize and write [B,S,E] bf16
  for (int r = 0; r < 8; r++) {
    const int m = r + 8 * g;
    const float inv = 1.0f / lrun[r];
    const int sIdx = q0 + wave * 16 + m;
    for (int nt = 0; nt < 4; nt++) {
      const int e = h * DD + nt * 16 + lr;
      attn_out[((size_t)b * SS + sIdx) * EE + e] = f2bf(o[nt][r] * inv);
    }
  }
}

// ---------------- host-side orchestration ----------------
extern "C" void kernel_launch(void* const* d_in, const int* in_sizes, int n_in,
                              void* d_out, int out_size, void* d_ws, size_t ws_size,
                              hipStream_t stream) {
  const float* q     = (const float*)d_in[0];
  const float* k     = (const float*)d_in[1];
  const float* v     = (const float*)d_in[2];
  const unsigned char* mask = (const unsigned char*)d_in[3];
  const float* W1 = (const float*)d_in[4];
  const float* b1 = (const float*)d_in[5];
  const float* W2 = (const float*)d_in[6];
  const float* b2 = (const float*)d_in[7];
  const float* W3 = (const float*)d_in[8];
  const float* b3 = (const float*)d_in[9];
  const float* W4 = (const float*)d_in[10];
  const float* b4 = (const float*)d_in[11];
  const float* gamma = (const float*)d_in[12];
  const float* beta  = (const float*)d_in[13];
  float* out = (float*)d_out;

  char* ws = (char*)d_ws;
  size_t off = 0;
  auto take = [&](size_t bytes) -> char* {
    char* p = ws + off;
    off += (bytes + 255) & ~(size_t)255;
    return p;
  };
  const size_t nBSE = (size_t)MM * EE;
  unsigned short* q_bf  = (unsigned short*)take(nBSE * 2);
  unsigned short* k_bf  = (unsigned short*)take(nBSE * 2);
  float*          vn_f  = (float*)take(nBSE * 4);
  unsigned short* vn_bf = (unsigned short*)take(nBSE * 2);
  unsigned short* wT_bf[4];
  for (int i = 0; i < 4; i++) wT_bf[i] = (unsigned short*)take((size_t)EE * EE * 2);
  unsigned short* qh   = (unsigned short*)take(nBSE * 2);
  unsigned short* kh   = (unsigned short*)take(nBSE * 2);
  unsigned short* vhT  = (unsigned short*)take(nBSE * 2);
  unsigned short* attn = (unsigned short*)take(nBSE * 2);
  float*          proj = (float*)take(nBSE * 4);
  (void)ws_size; (void)in_sizes; (void)n_in; (void)out_size;

  // 1) conversions (activations elementwise; weights transposed to [N][K])
  {
    int n = (int)nBSE;
    int blocks = (n + 255) / 256;
    cvt_bf16_kernel<<<blocks, 256, 0, stream>>>(q, q_bf, n);
    cvt_bf16_kernel<<<blocks, 256, 0, stream>>>(k, k_bf, n);
    dim3 tg(EE / 32, EE / 32);
    cvt_wT_kernel<<<tg, 256, 0, stream>>>(W1, wT_bf[0]);
    cvt_wT_kernel<<<tg, 256, 0, stream>>>(W2, wT_bf[1]);
    cvt_wT_kernel<<<tg, 256, 0, stream>>>(W3, wT_bf[2]);
    cvt_wT_kernel<<<tg, 256, 0, stream>>>(W4, wT_bf[3]);
  }

  // 2) LayerNorm(v) -> residual fp32 + bf16 copy
  ln_kernel<<<MM, 256, 0, stream>>>(v, gamma, beta, vn_f, vn_bf);

  // 3) projections: q,k -> [B,H,S,D]; v -> [B,H,D,S] (transposed for flash)
  dim3 gg(EE / 128, MM / 128);   // (8, 64) blocks of 128x128
  gemm_kernel<<<gg, 128, 0, stream>>>(q_bf,  wT_bf[0], b1, EE, EE, 0, qh,  nullptr, nullptr);
  gemm_kernel<<<gg, 128, 0, stream>>>(k_bf,  wT_bf[1], b2, EE, EE, 0, kh,  nullptr, nullptr);
  gemm_kernel<<<gg, 128, 0, stream>>>(vn_bf, wT_bf[2], b3, EE, EE, 2, vhT, nullptr, nullptr);

  // 4) flash attention
  flash_kernel<<<dim3(SS / 64, HH, BB), 128, 0, stream>>>(qh, kh, vhT, mask, attn);

  // 5) output projection + bias + residual (fp32)
  gemm_kernel<<<gg, 128, 0, stream>>>(attn, wT_bf[3], b4, EE, EE, 1, nullptr, proj, vn_f);

  // 6) final LayerNorm -> d_out
  ln_kernel<<<MM, 256, 0, stream>>>(proj, gamma, beta, out, nullptr);
}
